// S5SSM_65541200936977
// MI455X (gfx1250) — matrix operand order repaired and checked by reading, not compiled, not run on hardware
//
#include <hip/hip_runtime.h>
#include <hip/hip_bf16.h>

typedef __attribute__((ext_vector_type(16))) __bf16 v16bf;
typedef __attribute__((ext_vector_type(8)))  __bf16 v8bf;
typedef __attribute__((ext_vector_type(8)))  float  v8f;
typedef __attribute__((ext_vector_type(4)))  float  v4f;

#define BSZ   8
#define LSEQ  16384
#define HDIM  512
#define PDIM  256
#define N2    (2*PDIM)        // 512 packed cols (re|im) of GEMM1 / K of GEMM2
#define KDIM  512             // K for both GEMMs
#define NROWS (BSZ*LSEQ)      // 131072
#define LC    128             // scan chunk length
#define NCH   (LSEQ/LC)       // 128 chunks
#define BROWS 64              // rows per block
#define BCOLS 256             // cols per block
#define KSLICE 32             // K per pipeline stage
#define SLICE_BYTES (BCOLS*KSLICE*2)   // 16384 bytes per LDS buffer
#define WTILES 8              // col tiles per wave (16x128)

// ---------------------------------------------------------------------------
// Kernel 0: ZOH discretization + bf16 weight packing (K-contiguous):
//   W[n*KDIM + k];  Bbb: n = [re(p)|im(p)], k = h
//                   W2c: n = h, k = [2*C_re(p) | -2*C_im(p)]
//   lamAB[p*4] = {Lb_re, Lb_im, Lb^LC_re, Lb^LC_im}
// ---------------------------------------------------------------------------
__global__ void s5_setup(const float* __restrict__ Lre, const float* __restrict__ Lim,
                         const float* __restrict__ Bmat, const float* __restrict__ Cmat,
                         const float* __restrict__ logstep,
                         __bf16* __restrict__ Bbb, __bf16* __restrict__ W2c,
                         float* __restrict__ lamAB) {
    int p = blockIdx.x * blockDim.x + threadIdx.x;
    if (p >= PDIM) return;
    float lr = fminf(Lre[p], -1e-4f);
    float li = Lim[p];
    float st = __expf(logstep[p]);
    float zr = lr * st, zi = li * st;
    float er = __expf(zr);
    float lbr = er * __cosf(zi), lbi = er * __sinf(zi);     // Lambda_bar
    float nr = lbr - 1.0f, ni = lbi;                        // (Lb-1)/Lambda
    float den = lr * lr + li * li;
    float cr = (nr * lr + ni * li) / den;
    float ci = (ni * lr - nr * li) / den;
    for (int h = 0; h < HDIM; ++h) {
        float br = Bmat[((size_t)p * HDIM + h) * 2 + 0];
        float bi = Bmat[((size_t)p * HDIM + h) * 2 + 1];
        Bbb[(size_t)p * KDIM + h]          = (__bf16)(cr * br - ci * bi);
        Bbb[(size_t)(PDIM + p) * KDIM + h] = (__bf16)(cr * bi + ci * br);
        float CRe = Cmat[((size_t)h * PDIM + p) * 2 + 0];
        float CIm = Cmat[((size_t)h * PDIM + p) * 2 + 1];
        W2c[(size_t)h * KDIM + p]        = (__bf16)( 2.0f * CRe);
        W2c[(size_t)h * KDIM + PDIM + p] = (__bf16)(-2.0f * CIm);
    }
    float ar = lbr, ai = lbi;                               // Lb^128: 7 squarings
    #pragma unroll
    for (int s = 0; s < 7; ++s) { float t = ar * ar - ai * ai; ai = 2.f * ar * ai; ar = t; }
    lamAB[p * 4 + 0] = lbr; lamAB[p * 4 + 1] = lbi;
    lamAB[p * 4 + 2] = ar;  lamAB[p * 4 + 3] = ai;
}

// ---------------------------------------------------------------------------
// Async global->LDS stage: thread tid copies one column's 32-K slice (64B)
// into LDS buffer `buf` using gfx1250 async-to-LDS ops (ASYNCcnt).
// ---------------------------------------------------------------------------
__device__ __forceinline__ void stage_b_slice(const __bf16* __restrict__ W,
                                              int c0, int k0, int tid, int buf) {
    const __bf16* gsrc = W + (size_t)(c0 + tid) * KDIM + k0;   // 64B contiguous
    unsigned ldst = (unsigned)(buf * SLICE_BYTES + tid * 64);
    asm volatile("global_load_async_to_lds_b128 %0, %1, off"            :: "v"(ldst), "v"(gsrc) : "memory");
    asm volatile("global_load_async_to_lds_b128 %0, %1, off offset:16"  :: "v"(ldst), "v"(gsrc) : "memory");
    asm volatile("global_load_async_to_lds_b128 %0, %1, off offset:32"  :: "v"(ldst), "v"(gsrc) : "memory");
    asm volatile("global_load_async_to_lds_b128 %0, %1, off offset:48"  :: "v"(ldst), "v"(gsrc) : "memory");
}

// ---------------------------------------------------------------------------
// Fused WMMA GEMM: block = 64 rows x 256 cols; 8 waves = 4 row-strips x 2
// col-halves; wave = 16 rows x 8 col-tiles (64 acc VGPRs -> no VGPR-MSB
// traffic, high occupancy). B panel double-buffered in LDS via async copies.
// MODE 0: A = f32 (u), Out = Bu               (input projection)
// MODE 1: A = bf16 (xs), Out = ys + D[h]*u    (output projection+feedthrough)
// ---------------------------------------------------------------------------
template <int MODE>
__global__ __launch_bounds__(256) void s5_gemm_wmma(const void* __restrict__ Asrc,
                                                    const __bf16* __restrict__ W,
                                                    const float* __restrict__ u,
                                                    const float* __restrict__ Dvec,
                                                    float* __restrict__ Out) {
    extern __shared__ __bf16 sb[];            // 2 x 16KB double buffer
    const int tid  = threadIdx.x;
    const int lane = tid & 31;
    const int wave = tid >> 5;
    const int row0 = blockIdx.x * BROWS + (wave >> 1) * 16;
    const int chalf= (wave & 1) * (BCOLS / 2);            // 0 or 128
    const int c0   = blockIdx.y * BCOLS;
    const int m    = lane & 15;
    const int hi   = lane >> 4;
    const int aoff = hi * 16;                 // A-frag K run {aoff..aoff+15}
    v8f acc[WTILES] = {};

    stage_b_slice(W, c0, 0, tid, 0);

    const float*  arowf = (const float*)Asrc  + (size_t)(row0 + m) * KDIM;
    const __bf16* arowb = (const __bf16*)Asrc + (size_t)(row0 + m) * KDIM;

    for (int it = 0; it < KDIM / KSLICE; ++it) {
        const int k0  = it * KSLICE;
        const int cur = it & 1;
        asm volatile("s_wait_asynccnt 0x0" ::: "memory");   // staged slice landed
        asm volatile("s_wait_dscnt 0x0"    ::: "memory");   // my prev B reads done
        __syncthreads();
        if (it + 1 < KDIM / KSLICE)
            stage_b_slice(W, c0, k0 + KSLICE, tid, cur ^ 1);

        // A fragment (contiguous: lanes0-15 K=0..15, lanes16-31 K=16..31)
        v16bf a;
        if (MODE == 0) {
            __builtin_prefetch(arowf + k0 + 64, 0, 1);
            v4f a0 = *(const v4f*)(arowf + k0 + aoff);
            v4f a1 = *(const v4f*)(arowf + k0 + aoff + 4);
            v4f a2 = *(const v4f*)(arowf + k0 + aoff + 8);
            v4f a3 = *(const v4f*)(arowf + k0 + aoff + 12);
            #pragma unroll
            for (int e = 0; e < 4; ++e) {
                a[e]      = (__bf16)a0[e];
                a[4 + e]  = (__bf16)a1[e];
                a[8 + e]  = (__bf16)a2[e];
                a[12 + e] = (__bf16)a3[e];
            }
        } else {
            __builtin_prefetch(arowb + k0 + 64, 0, 1);
            v8bf a0 = *(const v8bf*)(arowb + k0 + aoff);
            v8bf a1 = *(const v8bf*)(arowb + k0 + aoff + 8);
            #pragma unroll
            for (int e = 0; e < 8; ++e) { a[e] = a0[e]; a[8 + e] = a1[e]; }
        }

        // 8 independent WMMAs reuse one A fragment; B frags from LDS (32B)
        const __bf16* sbase = sb + cur * (SLICE_BYTES / 2) + hi * 16;
        #pragma unroll
        for (int t = 0; t < WTILES; ++t) {
            v16bf b = *(const v16bf*)(sbase + (chalf + t * 16 + m) * KSLICE);
            acc[t] = __builtin_amdgcn_wmma_f32_16x16x32_bf16(false, a, false, b,
                                                             (short)0, acc[t], false, false);
        }
    }

    #pragma unroll
    for (int t = 0; t < WTILES; ++t) {
        int col = c0 + chalf + t * 16 + m;
        float dcoef = (MODE == 1) ? Dvec[col] : 0.0f;
        #pragma unroll
        for (int r = 0; r < 8; ++r) {
            int row = row0 + r + hi * 8;        // C/D layout: VGPR r -> M=r / r+8
            float v = acc[t][r];
            if (MODE == 1) v += dcoef * u[(size_t)row * HDIM + col];
            Out[(size_t)row * ((MODE == 0) ? N2 : HDIM) + col] = v;
        }
    }
}

// ---------------------------------------------------------------------------
// Scan phase 1: per (b, chunk, p) chunk-local end state (zero init); lanes map
// to consecutive p -> coalesced Bu reads.
// ---------------------------------------------------------------------------
__global__ __launch_bounds__(256) void s5_scan_local(const float* __restrict__ Bu,
                                                     const float* __restrict__ lamAB,
                                                     float* __restrict__ carry) {
    int tid = blockIdx.x * blockDim.x + threadIdx.x;   // BSZ*NCH*PDIM
    int p  = tid & (PDIM - 1);
    int ch = (tid >> 8) & (NCH - 1);
    int b  = tid >> 15;
    float ar = lamAB[p * 4 + 0], ai = lamAB[p * 4 + 1];
    const float* base = Bu + (size_t)(b * LSEQ + ch * LC) * N2;
    float xr = 0.f, xi = 0.f;
    for (int j = 0; j < LC; ++j) {
        float br = base[(size_t)j * N2 + p];
        float bi = base[(size_t)j * N2 + PDIM + p];
        float tr = ar * xr - ai * xi + br;
        xi       = ar * xi + ai * xr + bi;
        xr = tr;
    }
    size_t ci = ((size_t)(b * NCH + ch) * PDIM + p) * 2;
    carry[ci] = xr; carry[ci + 1] = xi;
}

// ---------------------------------------------------------------------------
// Scan phase 2: exclusive prefix over chunk ends: c_{k+1} = A^LC * c_k + e_k
// ---------------------------------------------------------------------------
__global__ __launch_bounds__(256) void s5_scan_chunks(float* __restrict__ carry,
                                                      const float* __restrict__ lamAB) {
    int tid = blockIdx.x * blockDim.x + threadIdx.x;   // BSZ*PDIM
    int p = tid & (PDIM - 1);
    int b = tid >> 8;
    float Ar = lamAB[p * 4 + 2], Ai = lamAB[p * 4 + 3];
    float cr = 0.f, ci_ = 0.f;
    for (int k = 0; k < NCH; ++k) {
        size_t idx = ((size_t)(b * NCH + k) * PDIM + p) * 2;
        float er = carry[idx], ei = carry[idx + 1];
        carry[idx] = cr; carry[idx + 1] = ci_;          // exclusive prefix
        float tr = Ar * cr - Ai * ci_ + er;
        ci_      = Ar * ci_ + Ai * cr + ei;
        cr = tr;
    }
}

// ---------------------------------------------------------------------------
// Scan phase 3: replay chunk seeded with carry; emit xs (bf16) for GEMM2
// ---------------------------------------------------------------------------
__global__ __launch_bounds__(256) void s5_scan_final(const float* __restrict__ Bu,
                                                     const float* __restrict__ lamAB,
                                                     const float* __restrict__ carry,
                                                     __bf16* __restrict__ xs) {
    int tid = blockIdx.x * blockDim.x + threadIdx.x;
    int p  = tid & (PDIM - 1);
    int ch = (tid >> 8) & (NCH - 1);
    int b  = tid >> 15;
    float ar = lamAB[p * 4 + 0], ai = lamAB[p * 4 + 1];
    size_t ci = ((size_t)(b * NCH + ch) * PDIM + p) * 2;
    float xr = carry[ci], xi = carry[ci + 1];
    const float* base  = Bu + (size_t)(b * LSEQ + ch * LC) * N2;
    __bf16*      obase = xs + (size_t)(b * LSEQ + ch * LC) * N2;
    for (int j = 0; j < LC; ++j) {
        float br = base[(size_t)j * N2 + p];
        float bi = base[(size_t)j * N2 + PDIM + p];
        float tr = ar * xr - ai * xi + br;
        xi       = ar * xi + ai * xr + bi;
        xr = tr;
        obase[(size_t)j * N2 + p]        = (__bf16)xr;
        obase[(size_t)j * N2 + PDIM + p] = (__bf16)xi;
    }
}

// ---------------------------------------------------------------------------
// Launch: setup -> gemm1 -> scan(local, chunks, final) -> gemm2
// ws: [0,512K) Bbb  [512K,1M) W2c  [1M,+4K) lamAB  [1028K,+2M) carry
//     [4M,+256M) Bu f32   [260M,+128M) xs bf16
// ---------------------------------------------------------------------------
extern "C" void kernel_launch(void* const* d_in, const int* in_sizes, int n_in,
                              void* d_out, int out_size, void* d_ws, size_t ws_size,
                              hipStream_t stream) {
    const float* u       = (const float*)d_in[0];
    const float* Lre     = (const float*)d_in[1];
    const float* Lim     = (const float*)d_in[2];
    const float* Bmat    = (const float*)d_in[3];
    const float* Cmat    = (const float*)d_in[4];
    const float* D       = (const float*)d_in[5];
    const float* logstep = (const float*)d_in[6];
    float* out = (float*)d_out;

    char* ws = (char*)d_ws;
    __bf16* Bbb   = (__bf16*)(ws);
    __bf16* W2c   = (__bf16*)(ws + (512u << 10));
    float*  lamAB = (float*) (ws + (1024u << 10));
    float*  carry = (float*) (ws + (1028u << 10));
    float*  Bu    = (float*) (ws + (4ull << 20));
    __bf16* xs    = (__bf16*)(ws + (4ull << 20) + (size_t)NROWS * N2 * sizeof(float));

    const size_t smem = 2 * SLICE_BYTES;   // 32 KB double buffer

    hipLaunchKernelGGL(s5_setup, dim3(1), dim3(256), 0, stream,
                       Lre, Lim, Bmat, Cmat, logstep, Bbb, W2c, lamAB);
    hipLaunchKernelGGL((s5_gemm_wmma<0>), dim3(NROWS / BROWS, N2 / BCOLS), dim3(256),
                       smem, stream, (const void*)u, Bbb, u, D, Bu);
    hipLaunchKernelGGL(s5_scan_local, dim3((BSZ * NCH * PDIM) / 256), dim3(256), 0, stream,
                       Bu, lamAB, carry);
    hipLaunchKernelGGL(s5_scan_chunks, dim3((BSZ * PDIM) / 256), dim3(256), 0, stream,
                       carry, lamAB);
    hipLaunchKernelGGL(s5_scan_final, dim3((BSZ * NCH * PDIM) / 256), dim3(256), 0, stream,
                       Bu, lamAB, carry, xs);
    hipLaunchKernelGGL((s5_gemm_wmma<1>), dim3(NROWS / BROWS, HDIM / BCOLS), dim3(256),
                       smem, stream, (const void*)xs, W2c, u, D, out);
}
